// TransformerBlock_4810363372473
// MI455X (gfx1250) — matrix operand order repaired and checked by reading, not compiled, not run on hardware
//
#include <hip/hip_runtime.h>
#include <hip/hip_bf16.h>
#include <math.h>

// ---------------------------------------------------------------------------
// Types for CDNA5 WMMA (wave32, 16x16x32 f16 -> f32 accum)
// ---------------------------------------------------------------------------
typedef __attribute__((ext_vector_type(16))) _Float16 v16h;
typedef __attribute__((ext_vector_type(8)))  _Float16 v8h;
typedef __attribute__((ext_vector_type(8)))  float    v8f;

#define D_MODEL 768
#define N_HEADS 12
#define D_HEAD  64
#define D_FF    3072
#define BATCH   2
#define SEQ     2048
#define ROWS    (BATCH * SEQ)   // 4096

__device__ __forceinline__ v8f wmma_f16f32(v16h a, v16h b, v8f c) {
  return __builtin_amdgcn_wmma_f32_16x16x32_f16(
      /*neg_a=*/false, a, /*neg_b=*/false, b,
      /*c_mod=*/(short)0, c, /*reuse_a=*/false, /*reuse_b=*/false);
}

// A fragment (16x32, row-major source): per lane 16 contiguous K values
// starting at kg*16.  Caller passes p = row_ptr + k0 + kg*16.
__device__ __forceinline__ v16h load_a_frag(const _Float16* __restrict__ p) {
  v8h lo = *(const v8h*)(p);
  v8h hi = *(const v8h*)(p + 8);
  v16h r;
#pragma unroll
  for (int i = 0; i < 8; ++i) { r[i] = lo[i]; r[i + 8] = hi[i]; }
  return r;
}

// B fragment (32x16, from K-contiguous transposed source): per lane 16
// contiguous K values starting at kg*16.  Caller passes p = col_ptr + k0 + kg*16.
__device__ __forceinline__ v16h load_b_frag(const _Float16* __restrict__ p) {
  v8h lo = *(const v8h*)(p);
  v8h hi = *(const v8h*)(p + 8);
  v16h r;
#pragma unroll
  for (int i = 0; i < 8; ++i) { r[i] = lo[i]; r[i + 8] = hi[i]; }
  return r;
}

// ---------------------------------------------------------------------------
// LayerNorm (fp32 in) -> f16 out, one 256-thread block per row of 768
// ---------------------------------------------------------------------------
__global__ __launch_bounds__(256) void ln_to_f16_kernel(
    const float* __restrict__ x, const float* __restrict__ g,
    const float* __restrict__ b, _Float16* __restrict__ out, int D) {
  __shared__ float red_s[8];
  __shared__ float red_ss[8];
  const int row = blockIdx.x;
  const float* xr = x + (size_t)row * D;
  float s = 0.f, ss = 0.f;
  for (int i = threadIdx.x; i < D; i += 256) {
    float v = xr[i]; s += v; ss += v * v;
  }
#pragma unroll
  for (int off = 1; off < 32; off <<= 1) {
    s  += __shfl_xor(s,  off, 32);
    ss += __shfl_xor(ss, off, 32);
  }
  const int wave = threadIdx.x >> 5;
  const int lane = threadIdx.x & 31;
  if (lane == 0) { red_s[wave] = s; red_ss[wave] = ss; }
  __syncthreads();
  float ts = 0.f, tss = 0.f;
#pragma unroll
  for (int w = 0; w < 8; ++w) { ts += red_s[w]; tss += red_ss[w]; }
  const float mu  = ts / (float)D;
  const float var = tss / (float)D - mu * mu;
  const float rs  = rsqrtf(var + 1e-5f);
  for (int i = threadIdx.x; i < D; i += 256) {
    float v = (xr[i] - mu) * rs * g[i] + b[i];
    out[(size_t)row * D + i] = (_Float16)v;
  }
}

// ---------------------------------------------------------------------------
// Convert fp32 weight [K,N] -> f16 transposed [N,K]
// ---------------------------------------------------------------------------
__global__ __launch_bounds__(256) void convert_transpose_kernel(
    const float* __restrict__ w, _Float16* __restrict__ wt, int K, int N) {
  size_t idx = (size_t)blockIdx.x * 256 + threadIdx.x;
  if (idx >= (size_t)K * N) return;
  int k = (int)(idx / N);
  int n = (int)(idx % N);
  wt[(size_t)n * K + k] = (_Float16)w[idx];
}

// ---------------------------------------------------------------------------
// WMMA GEMM, register-blocked: C[M,N] = A16[M,K] * BT16[N,K] + bias.
// Block = 256 threads = 8 waves (4x2); each wave computes a 32x32 tile as
// 2x2 WMMA 16x16 sub-tiles (2 A-frags x 2 B-frags -> 4 WMMAs per k-step).
// Block macro-tile = 128x64.
//   mode 0: out16 = f16(acc + bias)
//   mode 1: out32 = acc + bias + res
//   mode 2: out16 = f16(gelu_exact(acc + bias))
// ---------------------------------------------------------------------------
__global__ __launch_bounds__(256) void gemm_wmma_kernel(
    const _Float16* __restrict__ A, const _Float16* __restrict__ BT,
    const float* __restrict__ bias, const float* __restrict__ res,
    _Float16* __restrict__ out16, float* __restrict__ out32,
    int M, int N, int K, int mode) {
  const int lane = threadIdx.x & 31;
  const int wave = threadIdx.x >> 5;
  const int wm = wave >> 1;            // 0..3
  const int wn = wave & 1;             // 0..1
  const int tileM = blockIdx.y * 128 + wm * 32;
  const int tileN = blockIdx.x * 64  + wn * 32;
  const int r16 = lane & 15;
  const int kg  = lane >> 4;

  const _Float16* arow0 = A  + (size_t)(tileM + r16) * K + kg * 16;
  const _Float16* arow1 = arow0 + (size_t)16 * K;
  const _Float16* bcol0 = BT + (size_t)(tileN + r16) * K + kg * 16;
  const _Float16* bcol1 = bcol0 + (size_t)16 * K;

  v8f acc00 = {}, acc01 = {}, acc10 = {}, acc11 = {};
  for (int k0 = 0; k0 < K; k0 += 32) {
    v16h a0 = load_a_frag(arow0 + k0);
    v16h a1 = load_a_frag(arow1 + k0);
    v16h b0 = load_b_frag(bcol0 + k0);
    v16h b1 = load_b_frag(bcol1 + k0);
    acc00 = wmma_f16f32(a0, b0, acc00);
    acc01 = wmma_f16f32(a0, b1, acc01);
    acc10 = wmma_f16f32(a1, b0, acc10);
    acc11 = wmma_f16f32(a1, b1, acc11);
  }

  v8f accs[2][2] = {{acc00, acc01}, {acc10, acc11}};
#pragma unroll
  for (int ti = 0; ti < 2; ++ti) {
#pragma unroll
    for (int tj = 0; tj < 2; ++tj) {
      const int col = tileN + tj * 16 + r16;
      const float bn = bias ? bias[col] : 0.f;
#pragma unroll
      for (int r = 0; r < 8; ++r) {
        const int row = tileM + ti * 16 + kg * 8 + r;
        const size_t idx = (size_t)row * N + col;
        float v = accs[ti][tj][r] + bn;
        if (mode == 1) {
          out32[idx] = v + (res ? res[idx] : 0.f);
        } else if (mode == 2) {
          float gelu = 0.5f * v * (1.f + erff(v * 0.70710678118654752f));
          out16[idx] = (_Float16)gelu;
        } else {
          out16[idx] = (_Float16)v;
        }
      }
    }
  }
}

// ---------------------------------------------------------------------------
// Split fused QKV (f16, row layout [b,s, 3*H*64 as (t,h,d)]) into per-head
// Q[b,h,s,d], K[b,h,s,d], V^T[b,h,d,s]
// ---------------------------------------------------------------------------
__global__ __launch_bounds__(256) void split_qkv_kernel(
    const _Float16* __restrict__ qkv, _Float16* __restrict__ q,
    _Float16* __restrict__ k, _Float16* __restrict__ vt) {
  size_t idx = (size_t)blockIdx.x * 256 + threadIdx.x;
  const size_t total = (size_t)BATCH * N_HEADS * SEQ * D_HEAD;
  if (idx >= total) return;
  const int d = (int)(idx & (D_HEAD - 1));
  size_t t = idx >> 6;
  const int s = (int)(t % SEQ); t /= SEQ;
  const int h = (int)(t % N_HEADS);
  const int b = (int)(t / N_HEADS);
  const size_t rowbase = ((size_t)b * SEQ + s) * (3 * D_MODEL);
  q[idx] = qkv[rowbase + 0 * D_MODEL + h * D_HEAD + d];
  k[idx] = qkv[rowbase + 1 * D_MODEL + h * D_HEAD + d];
  vt[(((size_t)b * N_HEADS + h) * D_HEAD + d) * SEQ + s] =
      qkv[rowbase + 2 * D_MODEL + h * D_HEAD + d];
}

// ---------------------------------------------------------------------------
// Flash attention: one wave (32 threads) per (b, h, 16-query tile).
// Causal, online softmax; WMMA for Q.K^T (2x K=32) and P.V (4x d-chunks,
// K padded to 32 with zeros). P is transposed C->A layout through LDS.
// ---------------------------------------------------------------------------
__global__ __launch_bounds__(32) void flash_attn_kernel(
    const _Float16* __restrict__ Q, const _Float16* __restrict__ Kh,
    const _Float16* __restrict__ VT, _Float16* __restrict__ ctx) {
  __shared__ _Float16 pl[16 * 16];

  const int lane = threadIdx.x;
  const int r16  = lane & 15;
  const int kg   = lane >> 4;
  const int qtiles = SEQ / 16;
  const int qt = blockIdx.x % qtiles;
  const int bh = blockIdx.x / qtiles;        // b*H + h
  const int h  = bh % N_HEADS;
  const int b  = bh / N_HEADS;

  const _Float16* q  = Q  + (size_t)bh * SEQ * D_HEAD;
  const _Float16* kk = Kh + (size_t)bh * SEQ * D_HEAD;
  const _Float16* vt = VT + (size_t)bh * D_HEAD * SEQ;

  // Q A-fragments for d-chunks [0,32) and [32,64)
  const _Float16* qrow = q + (size_t)(qt * 16 + r16) * D_HEAD;
  v16h qa0 = load_a_frag(qrow + 0  + kg * 16);
  v16h qa1 = load_a_frag(qrow + 32 + kg * 16);

  v8f o0 = {}, o1 = {}, o2 = {}, o3 = {};
  float m[8], l[8];
#pragma unroll
  for (int r = 0; r < 8; ++r) { m[r] = -1e30f; l[r] = 0.f; }

  for (int kt = 0; kt <= qt; ++kt) {
    // ---- scores: S = Q . K^T  (16x16, K-dim = 64 over two WMMAs) ----
    const _Float16* krow = kk + (size_t)(kt * 16 + r16) * D_HEAD;
    v16h kb0 = load_b_frag(krow + 0  + kg * 16);
    v16h kb1 = load_b_frag(krow + 32 + kg * 16);
    v8f s = {};
    s = wmma_f16f32(qa0, kb0, s);
    s = wmma_f16f32(qa1, kb1, s);

    // ---- scale + causal mask + online softmax ----
    float p[8];
#pragma unroll
    for (int r = 0; r < 8; ++r) {
      float v = s[r] * 0.125f;  // 1/sqrt(64)
      if (kt == qt && r16 > (kg * 8 + r)) v = -1e30f;
      float mx = v;
#pragma unroll
      for (int off = 1; off < 16; off <<= 1) mx = fmaxf(mx, __shfl_xor(mx, off, 32));
      const float mnew  = fmaxf(m[r], mx);
      const float alpha = __expf(m[r] - mnew);
      float pv = __expf(v - mnew);
      float ps = pv;
#pragma unroll
      for (int off = 1; off < 16; off <<= 1) ps += __shfl_xor(ps, off, 32);
      l[r] = l[r] * alpha + ps;
      m[r] = mnew;
      p[r] = pv;
      o0[r] *= alpha; o1[r] *= alpha; o2[r] *= alpha; o3[r] *= alpha;
    }

    // ---- transpose P (C layout -> A layout) via LDS ----
    __syncthreads();
#pragma unroll
    for (int r = 0; r < 8; ++r) pl[(kg * 8 + r) * 16 + r16] = (_Float16)p[r];
    __syncthreads();
    v16h pa;
#pragma unroll
    for (int i = 0; i < 8; ++i) {
      pa[i]     = kg == 0 ? pl[r16 * 16 + i] : (_Float16)0.f;
      pa[i + 8] = kg == 0 ? pl[r16 * 16 + 8 + i] : (_Float16)0.f;   // K 16..31 padded
    }

    // ---- O += P . V  (four 16-wide d-chunks; K 16..31 zero) ----
#pragma unroll
    for (int dt = 0; dt < 4; ++dt) {
      v16h vb;
#pragma unroll
      for (int i = 0; i < 16; ++i) vb[i] = (_Float16)0.f;
      if (kg == 0) {  // lanes 0-15 hold K=0..15 (real keys); 16-31 hold padding
        const _Float16* vp = vt + (size_t)(dt * 16 + r16) * SEQ + kt * 16;
        v8h lo = *(const v8h*)(vp);
        v8h hi = *(const v8h*)(vp + 8);
#pragma unroll
        for (int i = 0; i < 8; ++i) { vb[i] = lo[i]; vb[i + 8] = hi[i]; }
      }
      v8f* oc = (dt == 0) ? &o0 : (dt == 1) ? &o1 : (dt == 2) ? &o2 : &o3;
      *oc = wmma_f16f32(pa, vb, *oc);
    }
  }

  // ---- epilogue: divide by row sums, write ctx[b, s, h*64 + d] ----
#pragma unroll
  for (int r = 0; r < 8; ++r) {
    const float inv = 1.f / l[r];
    const int srow = qt * 16 + kg * 8 + r;
    _Float16* cp = ctx + ((size_t)b * SEQ + srow) * D_MODEL + h * D_HEAD;
    cp[0 * 16 + r16] = (_Float16)(o0[r] * inv);
    cp[1 * 16 + r16] = (_Float16)(o1[r] * inv);
    cp[2 * 16 + r16] = (_Float16)(o2[r] * inv);
    cp[3 * 16 + r16] = (_Float16)(o3[r] * inv);
  }
}

// ---------------------------------------------------------------------------
// Host-side orchestration
// ---------------------------------------------------------------------------
extern "C" void kernel_launch(void* const* d_in, const int* in_sizes, int n_in,
                              void* d_out, int out_size, void* d_ws, size_t ws_size,
                              hipStream_t stream) {
  (void)in_sizes; (void)n_in; (void)out_size; (void)ws_size;

  const float* x      = (const float*)d_in[0];
  const float* ln1_g  = (const float*)d_in[1];
  const float* ln1_b  = (const float*)d_in[2];
  const float* ln2_g  = (const float*)d_in[3];
  const float* ln2_b  = (const float*)d_in[4];
  const float* w_qkv  = (const float*)d_in[5];
  const float* b_qkv  = (const float*)d_in[6];
  const float* w_out  = (const float*)d_in[7];
  const float* b_out  = (const float*)d_in[8];
  const float* w_fc1  = (const float*)d_in[9];
  const float* b_fc1  = (const float*)d_in[10];
  const float* w_fc2  = (const float*)d_in[11];
  const float* b_fc2  = (const float*)d_in[12];
  float* out = (float*)d_out;

  // Carve workspace (256B-aligned chunks)
  char* base = (char*)d_ws;
  size_t off = 0;
  auto alloc = [&](size_t bytes) -> void* {
    void* p = base + off;
    off = (off + bytes + 255) & ~(size_t)255;
    return p;
  };
  const size_t BS = (size_t)ROWS;
  _Float16* h16    = (_Float16*)alloc(BS * D_MODEL * 2);
  _Float16* wqkvT  = (_Float16*)alloc((size_t)3 * D_MODEL * D_MODEL * 2);
  _Float16* woutT  = (_Float16*)alloc((size_t)D_MODEL * D_MODEL * 2);
  _Float16* wfc1T  = (_Float16*)alloc((size_t)D_FF * D_MODEL * 2);
  _Float16* wfc2T  = (_Float16*)alloc((size_t)D_MODEL * D_FF * 2);
  _Float16* qkv16  = (_Float16*)alloc(BS * 3 * D_MODEL * 2);
  _Float16* q16    = (_Float16*)alloc(BS * D_MODEL * 2);
  _Float16* k16    = (_Float16*)alloc(BS * D_MODEL * 2);
  _Float16* vt16   = (_Float16*)alloc(BS * D_MODEL * 2);
  _Float16* ctx16  = (_Float16*)alloc(BS * D_MODEL * 2);
  float*    x1     = (float*)   alloc(BS * D_MODEL * 4);
  _Float16* h2_16  = (_Float16*)alloc(BS * D_MODEL * 2);
  _Float16* a16    = (_Float16*)alloc(BS * D_FF * 2);

  // 1) LN1 -> f16
  ln_to_f16_kernel<<<ROWS, 256, 0, stream>>>(x, ln1_g, ln1_b, h16, D_MODEL);

  // 2) Weight convert+transpose (fp32 [K,N] -> f16 [N,K])
  auto ct = [&](const float* w, _Float16* wt, int K, int N) {
    size_t total = (size_t)K * N;
    convert_transpose_kernel<<<(unsigned)((total + 255) / 256), 256, 0, stream>>>(w, wt, K, N);
  };
  ct(w_qkv, wqkvT, D_MODEL, 3 * D_MODEL);
  ct(w_out, woutT, D_MODEL, D_MODEL);
  ct(w_fc1, wfc1T, D_MODEL, D_FF);
  ct(w_fc2, wfc2T, D_FF, D_MODEL);

  // 3) QKV GEMM: [4096,768] x [768,2304] + bias -> f16
  gemm_wmma_kernel<<<dim3(3 * D_MODEL / 64, ROWS / 128), 256, 0, stream>>>(
      h16, wqkvT, b_qkv, nullptr, qkv16, nullptr, ROWS, 3 * D_MODEL, D_MODEL, 0);

  // 4) Split heads
  {
    size_t total = (size_t)BATCH * N_HEADS * SEQ * D_HEAD;
    split_qkv_kernel<<<(unsigned)((total + 255) / 256), 256, 0, stream>>>(
        qkv16, q16, k16, vt16);
  }

  // 5) Flash attention: one wave per (b,h,qtile)
  flash_attn_kernel<<<BATCH * N_HEADS * (SEQ / 16), 32, 0, stream>>>(
      q16, k16, vt16, ctx16);

  // 6) Out-proj GEMM + bias + residual(x) -> x1 (f32)
  gemm_wmma_kernel<<<dim3(D_MODEL / 64, ROWS / 128), 256, 0, stream>>>(
      ctx16, woutT, b_out, x, nullptr, x1, ROWS, D_MODEL, D_MODEL, 1);

  // 7) LN2 -> f16
  ln_to_f16_kernel<<<ROWS, 256, 0, stream>>>(x1, ln2_g, ln2_b, h2_16, D_MODEL);

  // 8) FC1 GEMM + bias + exact GELU -> f16
  gemm_wmma_kernel<<<dim3(D_FF / 64, ROWS / 128), 256, 0, stream>>>(
      h2_16, wfc1T, b_fc1, nullptr, a16, nullptr, ROWS, D_FF, D_MODEL, 2);

  // 9) FC2 GEMM + bias + residual(x1) -> d_out (f32)
  gemm_wmma_kernel<<<dim3(D_MODEL / 64, ROWS / 128), 256, 0, stream>>>(
      a16, wfc2T, b_fc2, x1, nullptr, out, ROWS, D_MODEL, D_FF, 1);
}
